// GAT_70093866271397
// MI455X (gfx1250) — matrix operand order repaired and checked by reading, not compiled, not run on hardware
//
#include <hip/hip_runtime.h>
#include <hip/hip_bf16.h>
#include <math.h>

typedef __bf16  v16bf __attribute__((ext_vector_type(16)));
typedef __bf16  v8bf  __attribute__((ext_vector_type(8)));
typedef float   v8f   __attribute__((ext_vector_type(8)));

#define GAT_N    30000
#define GAT_E    480000
#define GAT_FIN  4096
#define GAT_HC   512       // HEADS*HID
#define GAT_H    4
#define GAT_HID  128
#define GAT_CLS  6
#define NEG_SLOPE 0.2f

// ---------------------------------------------------------------------------
// helpers
// ---------------------------------------------------------------------------
__device__ __forceinline__ void atomicMaxF(float* a, float v) {
  // valid because segment-max buffers are initialized to -inf
  if (!(v < 0.0f)) atomicMax((int*)a, __float_as_int(v));
  else             atomicMin((unsigned int*)a, (unsigned int)__float_as_int(v));
}

__device__ __forceinline__ float wave_sum(float s) {
  #pragma unroll
  for (int off = 16; off; off >>= 1) s += __shfl_xor(s, off, 32);
  return s;
}

// LDS byte-offset of a pointer into a __shared__ object (addrspace(3) ptrtoint)
typedef __attribute__((address_space(3))) const void lds_cv;
__device__ __forceinline__ unsigned lds_addr_of(const void* p) {
  return (unsigned)(unsigned long long)(lds_cv*)p;
}

// per-lane async copy: 16B global -> 16B LDS, tracked by ASYNCcnt
__device__ __forceinline__ void async_b128(unsigned lds_off, const void* g) {
  asm volatile("global_load_async_to_lds_b128 %0, %1, off"
               :: "v"(lds_off), "v"((unsigned long long)g)
               : "memory");
}
__device__ __forceinline__ void wait_async4() {
  asm volatile("s_wait_asynccnt 0x4" ::: "memory");
}
__device__ __forceinline__ void wait_async0() {
  asm volatile("s_wait_asynccnt 0x0" ::: "memory");
}

// ---------------------------------------------------------------------------
// precision conversion passes (amortized: x is re-read 4x by the GEMM)
// ---------------------------------------------------------------------------
__global__ __launch_bounds__(256) void gat_cvt_bf16(
    const float* __restrict__ src, __bf16* __restrict__ dst, long total)
{
  long i = ((long)blockIdx.x * blockDim.x + threadIdx.x) * 8;
  if (i >= total) return;
  float4 a = *(const float4*)&src[i];
  float4 b = *(const float4*)&src[i + 4];
  v8bf o;
  o[0] = (__bf16)a.x; o[1] = (__bf16)a.y; o[2] = (__bf16)a.z; o[3] = (__bf16)a.w;
  o[4] = (__bf16)b.x; o[5] = (__bf16)b.y; o[6] = (__bf16)b.z; o[7] = (__bf16)b.w;
  *(v8bf*)&dst[i] = o;
}

// W1[4096,512] f32 -> W1T[512,4096] bf16 (4MB out, L2-resident)
__global__ __launch_bounds__(256) void gat_cvt_w1t(
    const float* __restrict__ W1, __bf16* __restrict__ W1T)
{
  int idx = blockIdx.x * 256 + threadIdx.x;
  if (idx >= GAT_FIN * GAT_HC) return;
  int k = idx >> 9;          // / 512
  int n = idx & 511;
  W1T[(size_t)n * GAT_FIN + k] = (__bf16)W1[idx];
}

// ---------------------------------------------------------------------------
// GEMM1: h1[30000,512] = x @ W1 in bf16 WMMA, f32 acc.
// Block tile 128x128, 8 waves (4M x 2N), wave tile 32x64 = 2x4 WMMA 16x16x32.
// A (xb, K-major) and B (W1T, K-major) tiles streamed global->LDS with
// global_load_async_to_lds_b128, double-buffered (static buffer select via
// manual 2x unroll + tail peel); s_wait_asynccnt pipelines tile k+1's DMA
// under tile k's WMMAs.
// ---------------------------------------------------------------------------
#define LDH 40                 // padded LDS row stride in halfs (80B = 20 banks)
#define BUFB (128 * LDH * 2)   // bytes per LDS buffer (10240)
#define NK   (GAT_FIN / 32)    // 128 k-steps

__global__ __launch_bounds__(256) void gat_gemm1_wmma(
    const __bf16* __restrict__ XB, const __bf16* __restrict__ W1T,
    float* __restrict__ H1, int M)
{
  __shared__ __bf16 As[2][128 * LDH];
  __shared__ __bf16 Bs[2][128 * LDH];

  const int tid   = threadIdx.x;
  const int lane  = tid & 31;
  const int wave  = tid >> 5;
  const int waveM = wave & 3;
  const int waveN = wave >> 2;
  const int m0 = blockIdx.y * 128;
  const int n0 = blockIdx.x * 128;

  // --- per-thread copy assignment: 2 x 16B chunks of A and of B -----------
  // chunk id 0..511 over a 128x32-half tile: row = id>>2, col = (id&3)*8 halfs
  unsigned aoff0[2], boff0[2];
  const __bf16* gA[2];
  const __bf16* gB[2];
  #pragma unroll
  for (int i = 0; i < 2; ++i) {
    int id = tid * 2 + i;
    int r  = id >> 2;
    int c  = (id & 3) * 8;
    aoff0[i] = lds_addr_of(&As[0][r * LDH + c]);
    boff0[i] = lds_addr_of(&Bs[0][r * LDH + c]);
    int gr = min(m0 + r, M - 1);          // clamp: no divergence, rows >= M unused
    gA[i] = XB  + (size_t)gr * GAT_FIN + c;
    gB[i] = W1T + (size_t)(n0 + r) * GAT_FIN + c;
  }

  v8f acc[2][4] = {};

  const int half = lane >> 4;
  const int mr   = lane & 15;
  const int kkA  = half * 8;      // A frag: K = (i<8?i:i+8) + half*8
  const int kkB  = half * 8;      // B frag: K = (i<8?i:i+8) + half*8

  // issue the 4 DMAs of one 32-wide k-tile into buffer `bufbase`, then
  // advance the global pointers to the next k-tile.
  auto issue = [&](unsigned bufbase) {
    #pragma unroll
    for (int i = 0; i < 2; ++i) {
      async_b128(aoff0[i] + bufbase, gA[i]);
      async_b128(boff0[i] + bufbase, gB[i]);
      gA[i] += 32;
      gB[i] += 32;
    }
  };

  // 8 WMMAs on buffer `buf` (call with literal 0/1 -> constant LDS offsets)
  auto compute = [&](const int buf) {
    v16bf afrag[2], bfrag[4];
    #pragma unroll
    for (int mt = 0; mt < 2; ++mt) {
      const __bf16* p = &As[buf][(waveM * 32 + mt * 16 + mr) * LDH + kkA];
      v8bf lo = *(const v8bf*)p;
      v8bf hi = *(const v8bf*)(p + 16);
      afrag[mt] = __builtin_shufflevector(lo, hi, 0,1,2,3,4,5,6,7,8,9,10,11,12,13,14,15);
    }
    #pragma unroll
    for (int nt = 0; nt < 4; ++nt) {
      const __bf16* p = &Bs[buf][(waveN * 64 + nt * 16 + mr) * LDH + kkB];
      v8bf lo = *(const v8bf*)p;
      v8bf hi = *(const v8bf*)(p + 16);
      bfrag[nt] = __builtin_shufflevector(lo, hi, 0,1,2,3,4,5,6,7,8,9,10,11,12,13,14,15);
    }
    #pragma unroll
    for (int mt = 0; mt < 2; ++mt)
      #pragma unroll
      for (int nt = 0; nt < 4; ++nt)
        acc[mt][nt] = __builtin_amdgcn_wmma_f32_16x16x32_bf16(
            false, afrag[mt], false, bfrag[nt], (short)0, acc[mt][nt], false, false);
  };

  // prologue: DMA tile 0 into buffer 0
  issue(0);

  // steady state: branch-free, static double buffering (2x manual unroll)
  #pragma unroll 1
  for (int kt = 0; kt < NK - 2; kt += 2) {
    issue(BUFB);  wait_async4();  __syncthreads();
    compute(0);
    __syncthreads();
    issue(0);     wait_async4();  __syncthreads();
    compute(1);
    __syncthreads();
  }
  // tail: last two k-steps
  issue(BUFB);  wait_async4();  __syncthreads();
  compute(0);
  __syncthreads();
  wait_async0();  __syncthreads();
  compute(1);

  // epilogue: VGPR r -> lanes0-15 M=r, lanes16-31 M=8+r; N = lane%16
  #pragma unroll
  for (int mt = 0; mt < 2; ++mt)
    #pragma unroll
    for (int nt = 0; nt < 4; ++nt) {
      int n = n0 + waveN * 64 + nt * 16 + mr;
      #pragma unroll
      for (int r = 0; r < 8; ++r) {
        int m = m0 + waveM * 32 + mt * 16 + r + half * 8;
        if (m < M) H1[(size_t)m * GAT_HC + n] = acc[mt][nt][r];
      }
    }
}

// ---------------------------------------------------------------------------
// per-node attention scalars: one wave per (node, head)
// ---------------------------------------------------------------------------
__global__ __launch_bounds__(256) void gat_alpha1(
    const float* __restrict__ H1, const float* __restrict__ a_src,
    const float* __restrict__ a_dst, float* __restrict__ as_o,
    float* __restrict__ ad_o, int Nn)
{
  int gw   = (blockIdx.x * blockDim.x + threadIdx.x) >> 5;
  int lane = threadIdx.x & 31;
  if (gw >= Nn * GAT_H) return;
  int n = gw >> 2, h = gw & 3;
  const float* row = H1 + (size_t)n * GAT_HC + h * GAT_HID;
  const float* av  = a_src + h * GAT_HID;
  const float* dv  = a_dst + h * GAT_HID;
  float s = 0.f, d = 0.f;
  #pragma unroll
  for (int c = lane; c < GAT_HID; c += 32) {
    float v = row[c]; s += v * av[c]; d += v * dv[c];
  }
  s = wave_sum(s); d = wave_sum(d);
  if (lane == 0) { as_o[gw] = s; ad_o[gw] = d; }
}

// ---------------------------------------------------------------------------
// edge phases, layer 1 (4 heads)
// ---------------------------------------------------------------------------
__global__ void gat_edge_max1(const int* __restrict__ ei,
                              const float* __restrict__ as1,
                              const float* __restrict__ ad1,
                              float* __restrict__ p1, float* __restrict__ m1)
{
  int e = blockIdx.x * blockDim.x + threadIdx.x;
  if (e >= GAT_E + GAT_N) return;
  int s, d;
  if (e < GAT_E) { s = ei[e]; d = ei[GAT_E + e]; } else { s = d = e - GAT_E; }
  #pragma unroll
  for (int h = 0; h < GAT_H; ++h) {
    float v = as1[s * GAT_H + h] + ad1[d * GAT_H + h];
    v = v > 0.f ? v : NEG_SLOPE * v;
    p1[(size_t)e * GAT_H + h] = v;
    atomicMaxF(&m1[d * GAT_H + h], v);
  }
}

__global__ void gat_edge_exp1(const int* __restrict__ ei,
                              const float* __restrict__ m1,
                              float* __restrict__ p1, float* __restrict__ dn1)
{
  int e = blockIdx.x * blockDim.x + threadIdx.x;
  if (e >= GAT_E + GAT_N) return;
  int d = (e < GAT_E) ? ei[GAT_E + e] : (e - GAT_E);
  #pragma unroll
  for (int h = 0; h < GAT_H; ++h) {
    float p = __expf(p1[(size_t)e * GAT_H + h] - m1[d * GAT_H + h]);
    p1[(size_t)e * GAT_H + h] = p;
    atomicAdd(&dn1[d * GAT_H + h], p);
  }
}

// one wave per edge: 16 features/lane, alphas broadcast via shfl
__global__ __launch_bounds__(256) void gat_edge_agg1(
    const int* __restrict__ ei, const float* __restrict__ p1,
    const float* __restrict__ dn1, const float* __restrict__ H1,
    float* __restrict__ agg)
{
  int lane = threadIdx.x & 31;
  int e = blockIdx.x * 8 + (threadIdx.x >> 5);
  if (e >= GAT_E + GAT_N) return;
  int s, d;
  if (e < GAT_E) { s = ei[e]; d = ei[GAT_E + e]; } else { s = d = e - GAT_E; }
  float a = 0.f;
  if (lane < GAT_H) a = p1[(size_t)e * GAT_H + lane] / dn1[d * GAT_H + lane];
  float al[GAT_H];
  #pragma unroll
  for (int h = 0; h < GAT_H; ++h) al[h] = __shfl(a, h, 32);
  const float* src = H1 + (size_t)s * GAT_HC;
  float* dst = agg + (size_t)d * GAT_HC;
  #pragma unroll
  for (int i = 0; i < 16; ++i) {
    int f = i * 32 + lane;
    atomicAdd(&dst[f], src[f] * al[i >> 2]);
  }
}

// ---------------------------------------------------------------------------
// layer-1 bias + ELU (in place: agg1 -> h2)
// ---------------------------------------------------------------------------
__global__ void gat_elu_bias(float* __restrict__ agg, const float* __restrict__ b1, int total)
{
  int i = blockIdx.x * blockDim.x + threadIdx.x;
  if (i >= total) return;
  float v = agg[i] + b1[i & (GAT_HC - 1)];
  agg[i] = v > 0.f ? v : (__expf(v) - 1.f);
}

// ---------------------------------------------------------------------------
// GEMM2 (N=6) + layer-2 attention scalars; one wave per node, W2 in LDS
// ---------------------------------------------------------------------------
__global__ __launch_bounds__(256) void gat_gemm2_alpha2(
    const float* __restrict__ H2, const float* __restrict__ W2,
    const float* __restrict__ aS, const float* __restrict__ aD,
    float* __restrict__ g2, float* __restrict__ as2, float* __restrict__ ad2, int Nn)
{
  __shared__ float Ws[GAT_HC * GAT_CLS];
  for (int i = threadIdx.x; i < GAT_HC * GAT_CLS; i += 256) Ws[i] = W2[i];
  __syncthreads();
  int lane = threadIdx.x & 31, wave = threadIdx.x >> 5;
  int n = blockIdx.x * 8 + wave;
  if (n >= Nn) return;
  float acc[GAT_CLS] = {0.f, 0.f, 0.f, 0.f, 0.f, 0.f};
  for (int k = lane; k < GAT_HC; k += 32) {
    float hv = H2[(size_t)n * GAT_HC + k];
    #pragma unroll
    for (int c = 0; c < GAT_CLS; ++c) acc[c] += hv * Ws[k * GAT_CLS + c];
  }
  #pragma unroll
  for (int c = 0; c < GAT_CLS; ++c) acc[c] = wave_sum(acc[c]);
  if (lane == 0) {
    float s = 0.f, d = 0.f;
    #pragma unroll
    for (int c = 0; c < GAT_CLS; ++c) {
      g2[n * GAT_CLS + c] = acc[c];
      s += acc[c] * aS[c];
      d += acc[c] * aD[c];
    }
    as2[n] = s; ad2[n] = d;
  }
}

// ---------------------------------------------------------------------------
// layer-2 edge phases (H=1, C=6)
// ---------------------------------------------------------------------------
__global__ void gat_edge_max2(const int* __restrict__ ei,
                              const float* __restrict__ as2,
                              const float* __restrict__ ad2,
                              float* __restrict__ p2, float* __restrict__ m2)
{
  int e = blockIdx.x * blockDim.x + threadIdx.x;
  if (e >= GAT_E + GAT_N) return;
  int s, d;
  if (e < GAT_E) { s = ei[e]; d = ei[GAT_E + e]; } else { s = d = e - GAT_E; }
  float v = as2[s] + ad2[d];
  v = v > 0.f ? v : NEG_SLOPE * v;
  p2[e] = v;
  atomicMaxF(&m2[d], v);
}

__global__ void gat_edge_exp2(const int* __restrict__ ei,
                              const float* __restrict__ m2,
                              float* __restrict__ p2, float* __restrict__ dn2)
{
  int e = blockIdx.x * blockDim.x + threadIdx.x;
  if (e >= GAT_E + GAT_N) return;
  int d = (e < GAT_E) ? ei[GAT_E + e] : (e - GAT_E);
  float p = __expf(p2[e] - m2[d]);
  p2[e] = p;
  atomicAdd(&dn2[d], p);
}

__global__ void gat_edge_agg2(const int* __restrict__ ei,
                              const float* __restrict__ p2,
                              const float* __restrict__ dn2,
                              const float* __restrict__ g2,
                              float* __restrict__ out)
{
  int e = blockIdx.x * blockDim.x + threadIdx.x;
  if (e >= GAT_E + GAT_N) return;
  int s, d;
  if (e < GAT_E) { s = ei[e]; d = ei[GAT_E + e]; } else { s = d = e - GAT_E; }
  float a = p2[e] / dn2[d];
  #pragma unroll
  for (int c = 0; c < GAT_CLS; ++c)
    atomicAdd(&out[d * GAT_CLS + c], g2[s * GAT_CLS + c] * a);
}

// ---------------------------------------------------------------------------
// init kernels
// ---------------------------------------------------------------------------
__global__ void gat_fill(float* p, float v, int n) {
  int i = blockIdx.x * blockDim.x + threadIdx.x;
  if (i < n) p[i] = v;
}
__global__ void gat_init_out(float* out, const float* b2, int n) {
  int i = blockIdx.x * blockDim.x + threadIdx.x;
  if (i < n) out[i] = b2[i % GAT_CLS];
}

// ---------------------------------------------------------------------------
// launch
// ---------------------------------------------------------------------------
extern "C" void kernel_launch(void* const* d_in, const int* in_sizes, int n_in,
                              void* d_out, int out_size, void* d_ws, size_t ws_size,
                              hipStream_t stream) {
  const float* x     = (const float*)d_in[0];
  const int*   ei    = (const int*)  d_in[1];
  const float* W1    = (const float*)d_in[2];
  const float* asrc1 = (const float*)d_in[3];
  const float* adst1 = (const float*)d_in[4];
  const float* b1    = (const float*)d_in[5];
  const float* W2    = (const float*)d_in[6];
  const float* asrc2 = (const float*)d_in[7];
  const float* adst2 = (const float*)d_in[8];
  const float* b2    = (const float*)d_in[9];
  float* out = (float*)d_out;

  const int ET = GAT_E + GAT_N;

  char* ws = (char*)d_ws;
  auto carve = [&](size_t bytes) -> void* {
    char* p = ws;
    ws += (bytes + 255) & ~(size_t)255;
    return (void*)p;
  };
  __bf16* xb  = (__bf16*)carve((size_t)GAT_N * GAT_FIN * 2);
  __bf16* w1t = (__bf16*)carve((size_t)GAT_FIN * GAT_HC * 2);
  float* h1   = (float*)carve((size_t)GAT_N * GAT_HC * 4);
  float* agg1 = (float*)carve((size_t)GAT_N * GAT_HC * 4);
  float* as1  = (float*)carve((size_t)GAT_N * GAT_H * 4);
  float* ad1  = (float*)carve((size_t)GAT_N * GAT_H * 4);
  float* m1   = (float*)carve((size_t)GAT_N * GAT_H * 4);
  float* dn1  = (float*)carve((size_t)GAT_N * GAT_H * 4);
  float* p1   = (float*)carve((size_t)ET * GAT_H * 4);
  float* g2   = (float*)carve((size_t)GAT_N * GAT_CLS * 4);
  float* as2  = (float*)carve((size_t)GAT_N * 4);
  float* ad2  = (float*)carve((size_t)GAT_N * 4);
  float* m2   = (float*)carve((size_t)GAT_N * 4);
  float* dn2  = (float*)carve((size_t)GAT_N * 4);
  float* p2   = (float*)carve((size_t)ET * 4);

  // ---- init ----
  hipMemsetAsync(agg1, 0, (size_t)GAT_N * GAT_HC * 4, stream);
  hipMemsetAsync(dn1, 0, (size_t)GAT_N * GAT_H * 4, stream);
  hipMemsetAsync(dn2, 0, (size_t)GAT_N * 4, stream);
  gat_fill<<<(GAT_N * GAT_H + 255) / 256, 256, 0, stream>>>(m1, -INFINITY, GAT_N * GAT_H);
  gat_fill<<<(GAT_N + 255) / 256, 256, 0, stream>>>(m2, -INFINITY, GAT_N);
  gat_init_out<<<(GAT_N * GAT_CLS + 255) / 256, 256, 0, stream>>>(out, b2, GAT_N * GAT_CLS);

  // ---- precision conversion ----
  long xtotal = (long)GAT_N * GAT_FIN;
  gat_cvt_bf16<<<(int)((xtotal / 8 + 255) / 256), 256, 0, stream>>>(x, xb, xtotal);
  gat_cvt_w1t<<<(GAT_FIN * GAT_HC + 255) / 256, 256, 0, stream>>>(W1, w1t);

  // ---- layer 1 ----
  dim3 g1grid(GAT_HC / 128, (GAT_N + 127) / 128);
  gat_gemm1_wmma<<<g1grid, 256, 0, stream>>>(xb, w1t, h1, GAT_N);

  gat_alpha1<<<(GAT_N * GAT_H * 32 + 255) / 256, 256, 0, stream>>>(h1, asrc1, adst1, as1, ad1, GAT_N);

  gat_edge_max1<<<(ET + 255) / 256, 256, 0, stream>>>(ei, as1, ad1, p1, m1);
  gat_edge_exp1<<<(ET + 255) / 256, 256, 0, stream>>>(ei, m1, p1, dn1);
  gat_edge_agg1<<<(ET + 7) / 8, 256, 0, stream>>>(ei, p1, dn1, h1, agg1);

  gat_elu_bias<<<((GAT_N * GAT_HC) + 255) / 256, 256, 0, stream>>>(agg1, b1, GAT_N * GAT_HC);

  // ---- layer 2 ----
  gat_gemm2_alpha2<<<(GAT_N + 7) / 8, 256, 0, stream>>>(agg1, W2, asrc2, adst2,
                                                        g2, as2, ad2, GAT_N);
  gat_edge_max2<<<(ET + 255) / 256, 256, 0, stream>>>(ei, as2, ad2, p2, m2);
  gat_edge_exp2<<<(ET + 255) / 256, 256, 0, stream>>>(ei, m2, p2, dn2);
  gat_edge_agg2<<<(ET + 255) / 256, 256, 0, stream>>>(ei, p2, dn2, g2, out);
}
